// MambaTransformerblock_36910948942033
// MI455X (gfx1250) — matrix operand order repaired and checked by reading, not compiled, not run on hardware
//
#include <hip/hip_runtime.h>

// ---------------------------------------------------------------------------
// MI455X / gfx1250 Mamba+MQA transformer block.
// bf16 WMMA (v_wmma_f32_16x16x32_bf16) everywhere, f32 accumulate, wave32.
// A-operand LDS fragments are contiguous 16B loads, B fragments per-element reads; f32->bf16 uses
// the native RTNE cast; edge tiles use branchless clamped loads (no exec-mask
// divergence).
// ---------------------------------------------------------------------------

typedef __attribute__((ext_vector_type(16))) __bf16 bf16x16;
typedef __attribute__((ext_vector_type(8)))  __bf16 bf16x8;
typedef __attribute__((ext_vector_type(8)))  float  f32x8;

#define CAT16(lo, hi) __builtin_shufflevector(lo, hi, 0,1,2,3,4,5,6,7,8,9,10,11,12,13,14,15)

#define LSEQ   2048
#define DMODEL 768
#define DIN    1536
#define NSTATE 16
#define DTRANK 48
#define NHEAD  12
#define HDIM   64
#define FFDIM  3072
#define QKVW   896      // 768 + 64 + 64
#define DBCW   80       // DT_RANK + 2*N_STATE

// Native RTNE f32->bf16 (lowers to gfx1250 packed bf16 cvt).
__device__ __forceinline__ __bf16 f2bf(float f) { return (__bf16)f; }

__device__ __forceinline__ bf16x8 pack8(float4 a, float4 b) {
  bf16x8 p;
  p[0] = f2bf(a.x); p[1] = f2bf(a.y); p[2] = f2bf(a.z); p[3] = f2bf(a.w);
  p[4] = f2bf(b.x); p[5] = f2bf(b.y); p[6] = f2bf(b.z); p[7] = f2bf(b.w);
  return p;
}
__device__ __forceinline__ bf16x8 pack8s(float4 a, float4 b, float s) {
  bf16x8 p;
  p[0] = f2bf(a.x * s); p[1] = f2bf(a.y * s); p[2] = f2bf(a.z * s); p[3] = f2bf(a.w * s);
  p[4] = f2bf(b.x * s); p[5] = f2bf(b.y * s); p[6] = f2bf(b.z * s); p[7] = f2bf(b.w * s);
  return p;
}
__device__ __forceinline__ float geluf(float x) {
  return 0.5f * x * (1.0f + tanhf(0.7978845608028654f * (x + 0.044715f * x * x * x)));
}
__device__ __forceinline__ float softplusf(float x) {
  return (x > 25.0f) ? x : logf(1.0f + expf(x));
}
__device__ __forceinline__ float siluf(float x) {
  return x / (1.0f + expf(-x));
}

// ---------------------------------------------------------------------------
// LayerNorm: one block (256 threads = 8 waves) per row of 768.
// ---------------------------------------------------------------------------
__global__ __launch_bounds__(256) void ln_kernel(const float* __restrict__ x,
                                                 const float* __restrict__ w,
                                                 const float* __restrict__ b,
                                                 float* __restrict__ y) {
  const int row = blockIdx.x;
  const int tid = threadIdx.x;
  const float* xr = x + (size_t)row * DMODEL;
  float v0 = xr[tid], v1 = xr[tid + 256], v2 = xr[tid + 512];
  float s = v0 + v1 + v2;
  float q = v0 * v0 + v1 * v1 + v2 * v2;
#pragma unroll
  for (int m = 16; m >= 1; m >>= 1) { s += __shfl_xor(s, m, 32); q += __shfl_xor(q, m, 32); }
  __shared__ float ss[8], qq[8];
  const int wave = tid >> 5;
  if ((tid & 31) == 0) { ss[wave] = s; qq[wave] = q; }
  __syncthreads();
  float ts = 0.f, tq = 0.f;
#pragma unroll
  for (int i = 0; i < 8; ++i) { ts += ss[i]; tq += qq[i]; }
  const float mu = ts * (1.0f / DMODEL);
  const float var = tq * (1.0f / DMODEL) - mu * mu;
  const float rstd = rsqrtf(var + 1e-5f);
  float* yr = y + (size_t)row * DMODEL;
  yr[tid]       = (v0 - mu) * rstd * w[tid]       + b[tid];
  yr[tid + 256] = (v1 - mu) * rstd * w[tid + 256] + b[tid + 256];
  yr[tid + 512] = (v2 - mu) * rstd * w[tid + 512] + b[tid + 512];
}

// ---------------------------------------------------------------------------
// C[M,N] = act(A[M,K] @ W[N,K]^T + bias) + resid.
// Block = 256 threads (8 waves); tile M=128, N=64; wave = 16 rows x 64 cols
// (4 accumulators, 4 WMMAs per 32-K slab). A staged row-major (K contig),
// W staged K-major; A fragment is two 16B loads, B is read per element (144B stride).
// Edge slabs (K=48 / N=80) use branchless clamped loads: all K here are
// multiples of 16 and segments are 16-(A)/8-(W) aligned, so each segment is
// fully valid or fully invalid -> zero via multiply, clamp via address select.
// ---------------------------------------------------------------------------
__global__ __launch_bounds__(256) void gemm_wmma_kernel(const float* __restrict__ A, int lda,
                                                        const float* __restrict__ W, int ldw,
                                                        const float* __restrict__ bias,
                                                        const float* __restrict__ resid,
                                                        float* __restrict__ C, int ldc,
                                                        int N, int K, int act) {
  __shared__ alignas(16) __bf16 As[128][40];   // [m][k], pitch 80B
  __shared__ alignas(16) __bf16 Wt[32][72];    // [k][n], pitch 144B
  const int mBase = blockIdx.y * 128;
  const int nBase = blockIdx.x * 64;
  const int tid = threadIdx.x, lane = tid & 31, wave = tid >> 5;
  const int cl = lane & 15, hh = lane >> 4;
  const int wr = wave * 16;

  const int ar_ld = tid >> 1;                // staging: A row 0..127
  const int ac_ld = (tid & 1) * 16;          // A col seg
  const float* Arow0 = A + (size_t)(mBase + ar_ld) * lda;
  const int wn_ld = tid >> 2;                // staging: W row (n) 0..63
  const int wk_ld = (tid & 3) * 8;           // W col seg (k)
  const int nIdx = nBase + wn_ld;
  const int nClamp = (nIdx < N) ? nIdx : (N - 1);
  const float* WrowC = W + (size_t)nClamp * ldw;

  f32x8 acc[4] = {};

  for (int k0 = 0; k0 < K; k0 += 32) {
    __syncthreads();
    // ---- stage A (128x32) ----
    if (k0 + 32 <= K) {
      const float* ap = Arow0 + k0 + ac_ld;
      float4 f0 = *(const float4*)(ap);
      float4 f1 = *(const float4*)(ap + 4);
      float4 f2 = *(const float4*)(ap + 8);
      float4 f3 = *(const float4*)(ap + 12);
      *(bf16x8*)&As[ar_ld][ac_ld]     = pack8(f0, f1);
      *(bf16x8*)&As[ar_ld][ac_ld + 8] = pack8(f2, f3);
      if (k0 + 32 < K) __builtin_prefetch(ap + 32, 0, 0);
    } else {
      const int ka = k0 + ac_ld;
      const float sA = (ka < K) ? 1.0f : 0.0f;
      const float* ap = Arow0 + ((ka < K) ? ka : 0);
      float4 f0 = *(const float4*)(ap);
      float4 f1 = *(const float4*)(ap + 4);
      float4 f2 = *(const float4*)(ap + 8);
      float4 f3 = *(const float4*)(ap + 12);
      *(bf16x8*)&As[ar_ld][ac_ld]     = pack8s(f0, f1, sA);
      *(bf16x8*)&As[ar_ld][ac_ld + 8] = pack8s(f2, f3, sA);
    }
    // ---- stage W transposed (32k x 64n) ----
    if (k0 + 32 <= K && nBase + 64 <= N) {
      const float* wp = WrowC + k0 + wk_ld;
      float4 g0 = *(const float4*)(wp);
      float4 g1 = *(const float4*)(wp + 4);
      Wt[wk_ld + 0][wn_ld] = f2bf(g0.x);
      Wt[wk_ld + 1][wn_ld] = f2bf(g0.y);
      Wt[wk_ld + 2][wn_ld] = f2bf(g0.z);
      Wt[wk_ld + 3][wn_ld] = f2bf(g0.w);
      Wt[wk_ld + 4][wn_ld] = f2bf(g1.x);
      Wt[wk_ld + 5][wn_ld] = f2bf(g1.y);
      Wt[wk_ld + 6][wn_ld] = f2bf(g1.z);
      Wt[wk_ld + 7][wn_ld] = f2bf(g1.w);
      if (k0 + 32 < K) __builtin_prefetch(wp + 32, 0, 0);
    } else {
      const int kw = k0 + wk_ld;
      const float sW = (kw < K && nIdx < N) ? 1.0f : 0.0f;
      const float* wp = WrowC + ((kw < K) ? kw : 0);
      float4 g0 = *(const float4*)(wp);
      float4 g1 = *(const float4*)(wp + 4);
      Wt[wk_ld + 0][wn_ld] = f2bf(g0.x * sW);
      Wt[wk_ld + 1][wn_ld] = f2bf(g0.y * sW);
      Wt[wk_ld + 2][wn_ld] = f2bf(g0.z * sW);
      Wt[wk_ld + 3][wn_ld] = f2bf(g0.w * sW);
      Wt[wk_ld + 4][wn_ld] = f2bf(g1.x * sW);
      Wt[wk_ld + 5][wn_ld] = f2bf(g1.y * sW);
      Wt[wk_ld + 6][wn_ld] = f2bf(g1.z * sW);
      Wt[wk_ld + 7][wn_ld] = f2bf(g1.w * sW);
    }
    __syncthreads();

    // ---- fragments + 4 WMMAs ----
    const int kb = hh * 8;
    bf16x8 alo = *(const bf16x8*)&As[wr + cl][kb];
    bf16x8 ahi = *(const bf16x8*)&As[wr + cl][kb + 16];
    bf16x16 afr = CAT16(alo, ahi);
#pragma unroll
    for (int t = 0; t < 4; ++t) {
      bf16x16 bfr;
#pragma unroll
      for (int s = 0; s < 8; ++s) { bfr[s] = Wt[kb + s][t * 16 + cl]; bfr[s + 8] = Wt[kb + 16 + s][t * 16 + cl]; }
      acc[t] = __builtin_amdgcn_wmma_f32_16x16x32_bf16(false, afr, false, bfr, (short)0, acc[t], false, false);
    }
  }

  // C/D layout: row = 8*(lane>>4)+e, col = lane&15
  const int r0 = mBase + wr + hh * 8;
#pragma unroll
  for (int t = 0; t < 4; ++t) {
    const int col = nBase + t * 16 + cl;
    if (col < N) {
      const float bz = bias ? bias[col] : 0.0f;
#pragma unroll
      for (int e = 0; e < 8; ++e) {
        const int row = r0 + e;
        float v = acc[t][e] + bz;
        if (act == 1)      v = geluf(v);
        else if (act == 2) v = softplusf(v);
        if (resid) v += resid[(size_t)row * ldc + col];
        C[(size_t)row * ldc + col] = v;
      }
    }
  }
}

// ---------------------------------------------------------------------------
// Depthwise causal conv (D_CONV=4) + bias + SiLU. xm lives in big[:, :1536].
// ---------------------------------------------------------------------------
__global__ __launch_bounds__(256) void conv_silu_kernel(const float* __restrict__ big,
                                                        const float* __restrict__ cw,
                                                        const float* __restrict__ cb,
                                                        float* __restrict__ out) {
  const int idx = blockIdx.x * 256 + threadIdx.x;
  if (idx >= LSEQ * DIN) return;
  const int t = idx / DIN, c = idx - t * DIN;
  float acc = cb[c];
#pragma unroll
  for (int j = 0; j < 4; ++j) {
    const int tt = t - 3 + j;
    if (tt >= 0) acc += cw[c * 4 + j] * big[(size_t)tt * FFDIM + c];
  }
  out[(size_t)t * DIN + c] = siluf(acc);
}

// ---------------------------------------------------------------------------
// Selective scan + D skip + SiLU gate. 16 lanes per channel (one per state).
// ---------------------------------------------------------------------------
__global__ __launch_bounds__(256) void scan_gate_kernel(const float* __restrict__ delta,
                                                        const float* __restrict__ dbc,
                                                        const float* __restrict__ big,
                                                        float* __restrict__ xm,
                                                        const float* __restrict__ Alog,
                                                        const float* __restrict__ Dp) {
  const int lane = threadIdx.x & 15;                 // state n
  const int ch = blockIdx.x * 16 + (threadIdx.x >> 4);
  const float An = -expf(Alog[ch * NSTATE + lane]);
  const float Dc = Dp[ch];
  float h = 0.f;
  for (int t = 0; t < LSEQ; ++t) {
    const float dl = delta[(size_t)t * DIN + ch];
    const float xv = xm[(size_t)t * DIN + ch];
    const float Bv = dbc[(size_t)t * DBCW + DTRANK + lane];
    const float Cv = dbc[(size_t)t * DBCW + DTRANK + NSTATE + lane];
    h = expf(dl * An) * h + dl * Bv * xv;
    float p = h * Cv;
    p += __shfl_xor(p, 1, 16);
    p += __shfl_xor(p, 2, 16);
    p += __shfl_xor(p, 4, 16);
    p += __shfl_xor(p, 8, 16);
    const float res = big[(size_t)t * FFDIM + DIN + ch];
    const float g = (p + xv * Dc) * siluf(res);
    if (lane == 0) xm[(size_t)t * DIN + ch] = g;
  }
}

// ---------------------------------------------------------------------------
// Causal MQA flash attention. Grid (L/64, HEADS), 128 threads (4 waves),
// wave = 16 query rows. Computes S^T = K @ Q^T so that:
//   * A-operand (K tile) loads contiguous 16B; B-operand (Q^T, staged once) is read per element;
//   * the S^T accumulator layout (lane = qrow, elements = keys) is exactly
//     the A-fragment layout for P @ V — P never leaves registers;
//   * softmax stats are per-lane scalars (one shfl_xor(16) partner combine).
// ---------------------------------------------------------------------------
__global__ __launch_bounds__(128) void attn_flash_kernel(const float* __restrict__ qkv,
                                                         float* __restrict__ out) {
  __shared__ alignas(16) __bf16 Qt[64][72];   // [dim][qrow], pitch 144B
  __shared__ alignas(16) __bf16 Ks[64][72];   // [key][dim]
  __shared__ alignas(16) __bf16 Vs[64][72];   // [key][dim]

  const int qBase = blockIdx.x * 64;
  const int head = blockIdx.y;
  const int tid = threadIdx.x, lane = tid & 31, wave = tid >> 5;
  const int cl = lane & 15, hh = lane >> 4;
  const int halfrow = hh * 8;
  const int wr = wave * 16;

  // stage Q transposed once (scaled by 1/sqrt(64))
  {
    const int r = tid >> 1;
    const int c0 = (tid & 1) * 32;
    const float* qp = qkv + (size_t)(qBase + r) * QKVW + head * HDIM + c0;
#pragma unroll
    for (int j = 0; j < 8; ++j) {
      float4 f = *(const float4*)(qp + 4 * j);
      Qt[c0 + 4 * j + 0][r] = f2bf(f.x * 0.125f);
      Qt[c0 + 4 * j + 1][r] = f2bf(f.y * 0.125f);
      Qt[c0 + 4 * j + 2][r] = f2bf(f.z * 0.125f);
      Qt[c0 + 4 * j + 3][r] = f2bf(f.w * 0.125f);
    }
  }

  float mrow = -3.0e38f, lrow = 0.f;
  f32x8 O[4] = {};
  const int grow = qBase + wr + cl;   // this lane's query row

  for (int kt = 0; kt <= (int)blockIdx.x; ++kt) {
    __syncthreads();
    const int kBase = kt * 64;
    {
      const int r = tid >> 1;
      const int c0 = (tid & 1) * 32;
      const float* kp = qkv + (size_t)(kBase + r) * QKVW + DMODEL + c0;
      const float* vp = kp + HDIM;
#pragma unroll
      for (int j = 0; j < 4; ++j) {
        float4 a = *(const float4*)(kp + 8 * j);
        float4 b = *(const float4*)(kp + 8 * j + 4);
        *(bf16x8*)&Ks[r][c0 + 8 * j] = pack8(a, b);
        float4 c = *(const float4*)(vp + 8 * j);
        float4 d = *(const float4*)(vp + 8 * j + 4);
        *(bf16x8*)&Vs[r][c0 + 8 * j] = pack8(c, d);
      }
    }
    __syncthreads();

    // S^T = K @ Q^T : 4 key tiles x (16 qrows of this wave), contraction = 64 dims
    f32x8 S[4] = {};
#pragma unroll
    for (int kc = 0; kc < 64; kc += 32) {
      bf16x16 bq;
#pragma unroll
      for (int s = 0; s < 8; ++s) { bq[s] = Qt[kc + hh * 8 + s][wr + cl]; bq[s + 8] = Qt[kc + 16 + hh * 8 + s][wr + cl]; }
#pragma unroll
      for (int t = 0; t < 4; ++t) {
        bf16x8 al = *(const bf16x8*)&Ks[t * 16 + cl][hh * 8 + kc];
        bf16x8 ah = *(const bf16x8*)&Ks[t * 16 + cl][hh * 8 + kc + 16];
        bf16x16 ak = CAT16(al, ah);
        S[t] = __builtin_amdgcn_wmma_f32_16x16x32_bf16(false, ak, false, bq, (short)0, S[t], false, false);
      }
    }

    // online softmax over keys (rows of S^T live in this lane's elements)
    const bool diag = (kt == (int)blockIdx.x);
    float mx = mrow;
#pragma unroll
    for (int t = 0; t < 4; ++t)
#pragma unroll
      for (int e = 0; e < 8; ++e) {
        const int key = kBase + t * 16 + halfrow + e;
        if (diag && key > grow) S[t][e] = -3.0e38f;
        mx = fmaxf(mx, S[t][e]);
      }
    mx = fmaxf(mx, __shfl_xor(mx, 16, 32));
    const float corr = expf(mrow - mx);
    mrow = mx;
    float rs = 0.f;
    bf16x16 P01, P23;
#pragma unroll
    for (int e = 0; e < 8; ++e) {
      float p0 = expf(S[0][e] - mx);
      float p1 = expf(S[1][e] - mx);
      float p2 = expf(S[2][e] - mx);
      float p3 = expf(S[3][e] - mx);
      rs += (p0 + p1) + (p2 + p3);
      P01[e] = f2bf(p0); P01[e + 8] = f2bf(p1);
      P23[e] = f2bf(p2); P23[e + 8] = f2bf(p3);
    }
    rs += __shfl_xor(rs, 16, 32);
    lrow = lrow * corr + rs;
    // rescale O: per-row factor broadcast from lane (halfrow+e) of the 16-group
#pragma unroll
    for (int e = 0; e < 8; ++e) {
      const float ce = __shfl(corr, halfrow + e, 16);
      O[0][e] *= ce; O[1][e] *= ce; O[2][e] *= ce; O[3][e] *= ce;
    }
    // O += P @ V : P already in A-fragment layout (P01 = keys 0..31, P23 = 32..63)
#pragma unroll
    for (int kc = 0; kc < 64; kc += 32) {
      bf16x16 ap = (kc == 0) ? P01 : P23;
#pragma unroll
      for (int t = 0; t < 4; ++t) {
        bf16x16 bv;
#pragma unroll
        for (int s = 0; s < 8; ++s) { bv[s] = Vs[kc + halfrow + s][t * 16 + cl]; bv[s + 8] = Vs[kc + 16 + halfrow + s][t * 16 + cl]; }
        O[t] = __builtin_amdgcn_wmma_f32_16x16x32_bf16(false, ap, false, bv, (short)0, O[t], false, false);
      }
    }
  }

  const float invl = 1.0f / lrow;
#pragma unroll
  for (int e = 0; e < 8; ++e) {
    const float ie = __shfl(invl, halfrow + e, 16);
    const int row = qBase + wr + halfrow + e;
    float* o = out + (size_t)row * DMODEL + head * HDIM + cl;
    o[0]  = O[0][e] * ie;
    o[16] = O[1][e] * ie;
    o[32] = O[2][e] * ie;
    o[48] = O[3][e] * ie;
  }
}

// ---------------------------------------------------------------------------
// Host orchestration
// ---------------------------------------------------------------------------
extern "C" void kernel_launch(void* const* d_in, const int* in_sizes, int n_in,
                              void* d_out, int out_size, void* d_ws, size_t ws_size,
                              hipStream_t stream) {
  (void)in_sizes; (void)n_in; (void)out_size; (void)ws_size;
  const float* x          = (const float*)d_in[0];
  const float* norm_w     = (const float*)d_in[1];
  const float* norm_b     = (const float*)d_in[2];
  const float* m_in_w     = (const float*)d_in[3];
  const float* m_conv_w   = (const float*)d_in[4];
  const float* m_conv_b   = (const float*)d_in[5];
  const float* m_xproj_w  = (const float*)d_in[6];
  const float* m_dt_w     = (const float*)d_in[7];
  const float* m_dt_b     = (const float*)d_in[8];
  const float* m_Alog     = (const float*)d_in[9];
  const float* m_D        = (const float*)d_in[10];
  const float* m_out_w    = (const float*)d_in[11];
  const float* wqkv_w     = (const float*)d_in[12];
  const float* wqkv_b     = (const float*)d_in[13];
  const float* attn_out_w = (const float*)d_in[14];
  const float* attn_out_b = (const float*)d_in[15];
  const float* attn_norm_w= (const float*)d_in[16];
  const float* attn_norm_b= (const float*)d_in[17];
  const float* tffn_w1    = (const float*)d_in[18];
  const float* tffn_b1    = (const float*)d_in[19];
  const float* tffn_w2    = (const float*)d_in[20];
  const float* tffn_b2    = (const float*)d_in[21];
  const float* ffn_w1     = (const float*)d_in[22];
  const float* ffn_b1     = (const float*)d_in[23];
  const float* ffn_w2     = (const float*)d_in[24];
  const float* ffn_b2     = (const float*)d_in[25];
  float* outp = (float*)d_out;

  float* p = (float*)d_ws;
  float* bLN  = p; p += (size_t)LSEQ * DMODEL;
  float* bH   = p; p += (size_t)LSEQ * DMODEL;
  float* bBIG = p; p += (size_t)LSEQ * FFDIM;
  float* bD1  = p; p += (size_t)LSEQ * DIN;
  float* bD2  = p; p += (size_t)LSEQ * DIN;
  float* bSM  = p; p += (size_t)LSEQ * DBCW;
  float* bQKV = p; p += (size_t)LSEQ * QKVW;
  float* bT1  = p; p += (size_t)LSEQ * DMODEL;
  float* bT2  = p; p += (size_t)LSEQ * DMODEL;

  const dim3 blk256(256), blk128(128);
  const int MB = LSEQ / 128;  // 16 row-blocks of 128

  // h = LN(x)
  ln_kernel<<<LSEQ, blk256, 0, stream>>>(x, norm_w, norm_b, bLN);
  // xr = h @ in_w^T  (2048 x 3072)
  gemm_wmma_kernel<<<dim3(FFDIM / 64, MB), blk256, 0, stream>>>(
      bLN, DMODEL, m_in_w, DMODEL, nullptr, nullptr, bBIG, FFDIM, FFDIM, DMODEL, 0);
  // xm = silu(conv(xm) + b)
  conv_silu_kernel<<<(LSEQ * DIN + 255) / 256, blk256, 0, stream>>>(bBIG, m_conv_w, m_conv_b, bD1);
  // dbc = xm @ xproj^T  (N = 80)
  gemm_wmma_kernel<<<dim3(2, MB), blk256, 0, stream>>>(
      bD1, DIN, m_xproj_w, DIN, nullptr, nullptr, bSM, DBCW, DBCW, DIN, 0);
  // delta = softplus(dt @ dt_w^T + dt_b)  (K = 48, A-stride 80)
  gemm_wmma_kernel<<<dim3(DIN / 64, MB), blk256, 0, stream>>>(
      bSM, DBCW, m_dt_w, DTRANK, m_dt_b, nullptr, bD2, DIN, DIN, DTRANK, 2);
  // selective scan + D skip + silu(res) gate (in place over xm)
  scan_gate_kernel<<<DIN / 16, blk256, 0, stream>>>(bD2, bSM, bBIG, bD1, m_Alog, m_D);
  // h = gated @ out_w^T + h_ln_residual
  gemm_wmma_kernel<<<dim3(DMODEL / 64, MB), blk256, 0, stream>>>(
      bD1, DIN, m_out_w, DIN, nullptr, bLN, bH, DMODEL, DMODEL, DIN, 0);
  // h = LN(h)
  ln_kernel<<<LSEQ, blk256, 0, stream>>>(bH, norm_w, norm_b, bLN);
  // qkv = h @ wqkv^T + b  (N = 896)
  gemm_wmma_kernel<<<dim3(QKVW / 64, MB), blk256, 0, stream>>>(
      bLN, DMODEL, wqkv_w, DMODEL, wqkv_b, nullptr, bQKV, QKVW, QKVW, DMODEL, 0);
  // o = causal MQA(qkv)
  attn_flash_kernel<<<dim3(LSEQ / 64, NHEAD), blk128, 0, stream>>>(bQKV, bT1);
  // a = o @ attn_out^T + b
  gemm_wmma_kernel<<<dim3(DMODEL / 64, MB), blk256, 0, stream>>>(
      bT1, DMODEL, attn_out_w, DMODEL, attn_out_b, nullptr, bT2, DMODEL, DMODEL, DMODEL, 0);
  // a = LN(a)
  ln_kernel<<<LSEQ, blk256, 0, stream>>>(bT2, attn_norm_w, attn_norm_b, bT1);
  // a = gelu(a @ tffn_w1^T + b1)
  gemm_wmma_kernel<<<dim3(FFDIM / 64, MB), blk256, 0, stream>>>(
      bT1, DMODEL, tffn_w1, DMODEL, tffn_b1, nullptr, bBIG, FFDIM, FFDIM, DMODEL, 1);
  // h = (a @ tffn_w2^T + b2) + h
  gemm_wmma_kernel<<<dim3(DMODEL / 64, MB), blk256, 0, stream>>>(
      bBIG, FFDIM, tffn_w2, FFDIM, tffn_b2, bH, bH, DMODEL, DMODEL, FFDIM, 0);
  // h = LN(h)
  ln_kernel<<<LSEQ, blk256, 0, stream>>>(bH, norm_w, norm_b, bLN);
  // f = gelu(h @ ffn_w1^T + b1)
  gemm_wmma_kernel<<<dim3(FFDIM / 64, MB), blk256, 0, stream>>>(
      bLN, DMODEL, ffn_w1, DMODEL, ffn_b1, nullptr, bBIG, FFDIM, FFDIM, DMODEL, 1);
  // out = (f @ ffn_w2^T + b2) + h
  gemm_wmma_kernel<<<dim3(DMODEL / 64, MB), blk256, 0, stream>>>(
      bBIG, FFDIM, ffn_w2, FFDIM, ffn_b2, bH, outp, DMODEL, DMODEL, FFDIM, 0);
}
